// biRNNModel_30150670418270
// MI455X (gfx1250) — compile-verified, matrix operand order checked by reading, not run on hardware
//
#include <hip/hip_runtime.h>
#include <hip/hip_bf16.h>

#define B_ 1024
#define T_ 365

typedef __bf16 v16bf __attribute__((ext_vector_type(16)));
typedef float  v8f   __attribute__((ext_vector_type(8)));
typedef int    v4i   __attribute__((ext_vector_type(4)));

union Frag16 { uint4 u4[2]; v16bf v; };

// ---- CDNA5 async global->LDS path (guarded; falls back to sync copy) ----
#if defined(__HIP_DEVICE_COMPILE__) && __has_builtin(__builtin_amdgcn_global_load_async_to_lds_b128)
#define HAVE_ASYNC_LDS 1
#else
#define HAVE_ASYNC_LDS 0
#endif

__device__ __forceinline__ void copy16B_to_lds(unsigned short* dst_lds,
                                               const unsigned short* src_glb) {
#if HAVE_ASYNC_LDS
    __builtin_amdgcn_global_load_async_to_lds_b128(
        (__attribute__((address_space(1))) v4i*)(uintptr_t)src_glb,
        (__attribute__((address_space(3))) v4i*)(unsigned int)(uintptr_t)dst_lds,
        0, 0);
#else
    *(uint4*)dst_lds = *(const uint4*)src_glb;
#endif
}

__device__ __forceinline__ void wait_async_all() {
#if HAVE_ASYNC_LDS
#if __has_builtin(__builtin_amdgcn_s_wait_asynccnt)
    __builtin_amdgcn_s_wait_asynccnt(0);
#else
    asm volatile("s_wait_asynccnt 0x0" ::: "memory");
#endif
#endif
}

__device__ __forceinline__ unsigned short f2bf(float f) {
    union { float f; unsigned u; } x; x.f = f;
    unsigned r = (x.u + 0x7FFFu + ((x.u >> 16) & 1u)) >> 16;
    return (unsigned short)r;
}
__device__ __forceinline__ float bf2f(unsigned short b) {
    union { unsigned u; float f; } x; x.u = ((unsigned)b) << 16;
    return x.f;
}

// ---- shared A/B element index mapping (16-bit WMMA fragment layout) ----
__device__ __forceinline__ int frag_k(int lane, int e) {
    int g = lane >> 4;
    int i = e >> 1, h = e & 1;
    return (i < 4) ? (8 * g + 2 * i + h) : (16 + 8 * g + 2 * (i - 4) + h);
}

// ---- pack fp32 weight [K x 128] into WMMA B-tile fragments (bf16) ----
__global__ void pack_b_kernel(const float* __restrict__ W, int K,
                              unsigned short* __restrict__ out, int ntiles) {
    int tile = blockIdx.x;
    int nt = tile % ntiles;
    int kt = tile / ntiles;
    int lane = threadIdx.x >> 4;   // 0..31
    int e    = threadIdx.x & 15;   // 0..15
    int k = kt * 32 + frag_k(lane, e);
    int n = nt * 16 + (lane & 15);
    float v = (k < K) ? W[(size_t)k * (ntiles * 16) + n] : 0.0f;
    out[(size_t)tile * 512 + lane * 16 + e] = f2bf(v);
}

// ---- convert x [B,T,10] f32 -> [B,T,32] bf16 zero-padded ----
__global__ void pad_x_kernel(const float* __restrict__ x, unsigned short* __restrict__ xp) {
    size_t i = (size_t)blockIdx.x * 256 + threadIdx.x;
    if (i >= (size_t)B_ * T_ * 32) return;
    int k = (int)(i & 31);
    size_t bt = i >> 5;
    float v = (k < 10) ? x[bt * 10 + k] : 0.0f;
    xp[i] = f2bf(v);
}

// ---- gather a 16x32 A fragment from LDS (two 16B runs per lane) ----
__device__ __forceinline__ v16bf lds_a_frag(const unsigned short* s, int rowlen, int kt, int lane) {
    int l = lane & 15, g = lane >> 4;
    int k0 = kt * 32 + 8 * g;
    Frag16 f;
    f.u4[0] = *(const uint4*)(s + l * rowlen + k0);
    f.u4[1] = *(const uint4*)(s + l * rowlen + k0 + 16);
    return f.v;
}

// ---- stage one 16-row x K input tile into LDS (async if available) ----
template <int K>
__device__ __forceinline__ void stage_tile(unsigned short* __restrict__ dst,
                                           const unsigned short* __restrict__ in,
                                           int b0, int t, int tid) {
    constexpr int PER = K / 8;
    for (int c = tid; c < 16 * PER; c += 256) {
        int r = c / PER, cc = c % PER;
        copy16B_to_lds(&dst[r * K + cc * 8],
                       &in[((size_t)(b0 + r) * T_ + t) * K + cc * 8]);
    }
}

// ---- fused bidirectional SimpleRNN layer ----
// grid = (64, 2): 16 batch rows per block, blockIdx.y = direction.
// block = 256 threads = 8 waves; wave nt owns output columns [16nt,16nt+16).
// h_t = tanh(x_t @ wk + h_{t-1} @ wr + b); out = relu(h) into concat slot.
// Input tiles are double-buffered: step t+1 staged asynchronously (ASYNCcnt)
// while the WMMAs + tanh for step t execute.
template <int KT_IN>
__global__ __launch_bounds__(256) void scan_kernel(
    const unsigned short* __restrict__ in,   // [B, T, KT_IN*32] bf16
    const unsigned short* __restrict__ wkp_f, const unsigned short* __restrict__ wrp_f,
    const float* __restrict__ bias_f,
    const unsigned short* __restrict__ wkp_b, const unsigned short* __restrict__ wrp_b,
    const float* __restrict__ bias_b,
    unsigned short* __restrict__ out)        // [B, T, 256] bf16
{
    constexpr int K = KT_IN * 32;
    const int dir  = blockIdx.y;
    const int b0   = blockIdx.x * 16;
    const int tid  = threadIdx.x;
    const int lane = tid & 31;
    const int nt   = tid >> 5;
    const int l = lane & 15, g = lane >> 4;

    const unsigned short* wkp = dir ? wkp_b : wkp_f;
    const unsigned short* wrp = dir ? wrp_b : wrp_f;
    const float* bias = dir ? bias_b : bias_f;

    __shared__ unsigned short s_in[2][16 * K];
    __shared__ unsigned short s_h[16 * 128];

    // Preload all weight fragments into VGPRs (resident across the scan).
    Frag16 wk[KT_IN];
#pragma unroll
    for (int kt = 0; kt < KT_IN; ++kt) {
        const uint4* p = (const uint4*)(wkp + ((size_t)(kt * 8 + nt) * 512 + lane * 16));
        wk[kt].u4[0] = p[0]; wk[kt].u4[1] = p[1];
    }
    Frag16 wr[4];
#pragma unroll
    for (int kt = 0; kt < 4; ++kt) {
        const uint4* p = (const uint4*)(wrp + ((size_t)(kt * 8 + nt) * 512 + lane * 16));
        wr[kt].u4[0] = p[0]; wr[kt].u4[1] = p[1];
    }
    const float cb = bias[nt * 16 + l];

    for (int i = tid; i < 16 * 128; i += 256) s_h[i] = 0;

    // prime the pipeline: stage step 0
    stage_tile<K>(s_in[0], in, b0, dir ? (T_ - 1) : 0, tid);

    for (int step = 0; step < T_; ++step) {
        const int t   = dir ? (T_ - 1 - step) : step;
        const int cur = step & 1;

        wait_async_all();
        __syncthreads();   // s_in[cur] staged; prev step's s_h published

        // prefetch next step's input into the other buffer (overlaps compute)
        if (step + 1 < T_) {
            const int tn = dir ? (T_ - 2 - step) : (step + 1);
            stage_tile<K>(s_in[cur ^ 1], in, b0, tn, tid);
        }

        v8f c = {cb, cb, cb, cb, cb, cb, cb, cb};
#pragma unroll
        for (int kt = 0; kt < KT_IN; ++kt) {
            v16bf a = lds_a_frag(s_in[cur], K, kt, lane);
            c = __builtin_amdgcn_wmma_f32_16x16x32_bf16(false, a, false, wk[kt].v,
                                                        (short)0, c, false, false);
        }
#pragma unroll
        for (int kt = 0; kt < 4; ++kt) {
            v16bf a = lds_a_frag(s_h, 128, kt, lane);
            c = __builtin_amdgcn_wmma_f32_16x16x32_bf16(false, a, false, wr[kt].v,
                                                        (short)0, c, false, false);
        }

        float hv[8];
#pragma unroll
        for (int j = 0; j < 8; ++j) hv[j] = tanhf(c[j]);

        __syncthreads();   // all reads of old s_h complete
#pragma unroll
        for (int j = 0; j < 8; ++j) {
            int m = j + 8 * g;
            s_h[m * 128 + nt * 16 + l] = f2bf(hv[j]);
            out[(((size_t)(b0 + m)) * T_ + t) * 256 + dir * 128 + nt * 16 + l] =
                f2bf(fmaxf(hv[j], 0.0f));
        }
        // next iteration's first __syncthreads() publishes the new s_h
    }
}

// ---- stage a 16 x (ktn*32) A chunk for the dense GEMM ----
__device__ __forceinline__ void stage_dense(unsigned short* __restrict__ dst,
                                            const unsigned short* __restrict__ A,
                                            int b0, int K, int kc, int ktn, int tid) {
    int per = ktn * 4;  // 16B chunks per row
    for (int i = tid; i < 16 * per; i += 256) {
        int r = i / per, cc = i % per;
        copy16B_to_lds(&dst[r * 256 + cc * 8],
                       &A[(size_t)(b0 + r) * K + kc * 32 + cc * 8]);
    }
}

// ---- generic dense layer: out = relu(A @ W + b), N = 128 ----
// A: [M x K] bf16 row-major; Bp: packed B tiles [ktiles][8]; out bf16 [M x 128].
// A chunks double-buffered via async LDS loads; B fragments stream from L2.
__global__ __launch_bounds__(256) void dense_kernel(
    const unsigned short* __restrict__ A, int K, int ktiles,
    const unsigned short* __restrict__ Bp, const float* __restrict__ bias,
    unsigned short* __restrict__ out)
{
    const int b0 = blockIdx.x * 16;
    const int tid = threadIdx.x, lane = tid & 31, nt = tid >> 5;
    const int l = lane & 15, g = lane >> 4;
    __shared__ unsigned short s_a[2][16 * 256];

    const float cb = bias[nt * 16 + l];
    v8f c = {cb, cb, cb, cb, cb, cb, cb, cb};

    // prime: stage first K-chunk
    {
        int ktn0 = (ktiles < 8) ? ktiles : 8;
        stage_dense(s_a[0], A, b0, K, 0, ktn0, tid);
    }

    int buf = 0;
    for (int kc = 0; kc < ktiles; kc += 8, buf ^= 1) {
        int ktn = (ktiles - kc < 8) ? (ktiles - kc) : 8;

        wait_async_all();
        __syncthreads();   // s_a[buf] ready; prior reads of s_a[buf^1] done

        if (kc + 8 < ktiles) {
            int ktn_n = (ktiles - kc - 8 < 8) ? (ktiles - kc - 8) : 8;
            stage_dense(s_a[buf ^ 1], A, b0, K, kc + 8, ktn_n, tid);
        }

        for (int j = 0; j < ktn; ++j) {
            v16bf a = lds_a_frag(s_a[buf], 256, j, lane);
            Frag16 bf;
            const uint4* p = (const uint4*)(Bp + (((size_t)(kc + j) * 8 + nt) * 512 + lane * 16));
            bf.u4[0] = p[0]; bf.u4[1] = p[1];
            c = __builtin_amdgcn_wmma_f32_16x16x32_bf16(false, a, false, bf.v,
                                                        (short)0, c, false, false);
        }
    }
#pragma unroll
    for (int j = 0; j < 8; ++j) {
        int m = j + 8 * g;
        out[(size_t)(b0 + m) * 128 + nt * 16 + l] = f2bf(fmaxf(c[j], 0.0f));
    }
}

// ---- final GEMV: out[b] = h[b,:] . ow + ob ----
__global__ void head_kernel(const unsigned short* __restrict__ h,
                            const float* __restrict__ ow, const float* __restrict__ ob,
                            float* __restrict__ out) {
    int b = blockIdx.x * blockDim.x + threadIdx.x;
    if (b >= B_) return;
    float acc = ob[0];
    for (int u = 0; u < 128; ++u) acc += bf2f(h[(size_t)b * 128 + u]) * ow[u];
    out[b] = acc;
}

extern "C" void kernel_launch(void* const* d_in, const int* in_sizes, int n_in,
                              void* d_out, int out_size, void* d_ws, size_t ws_size,
                              hipStream_t stream) {
    const float* x = (const float*)d_in[0];
    const float *wk[3][2], *wr[3][2], *bs[3][2];
    int idx = 1;
    for (int layer = 0; layer < 3; ++layer)
        for (int d = 0; d < 2; ++d) {
            wk[layer][d] = (const float*)d_in[idx++];
            wr[layer][d] = (const float*)d_in[idx++];
            bs[layer][d] = (const float*)d_in[idx++];
        }
    const float* dw0 = (const float*)d_in[19];
    const float* db0 = (const float*)d_in[20];
    const float* dw1 = (const float*)d_in[21];
    const float* db1 = (const float*)d_in[22];
    const float* dw2 = (const float*)d_in[23];
    const float* db2 = (const float*)d_in[24];
    const float* ow  = (const float*)d_in[25];
    const float* ob  = (const float*)d_in[26];

    // bump allocator over d_ws
    char* base = (char*)d_ws;
    size_t off = 0;
    auto alloc = [&](size_t bytes) -> void* {
        void* p = base + off;
        off = (off + bytes + 255) & ~(size_t)255;
        return p;
    };

    unsigned short* xpad = (unsigned short*)alloc((size_t)B_ * T_ * 32 * 2);
    unsigned short* actA = (unsigned short*)alloc((size_t)B_ * T_ * 256 * 2);
    unsigned short* actB = (unsigned short*)alloc((size_t)B_ * T_ * 256 * 2);
    unsigned short *wkp[3][2], *wrp[3][2];
    const int kt_in[3] = {1, 8, 8};      // input K-tiles per layer (32 / 256 / 256)
    const int in_k[3]  = {10, 256, 256}; // true K per layer
    for (int layer = 0; layer < 3; ++layer)
        for (int d = 0; d < 2; ++d) {
            wkp[layer][d] = (unsigned short*)alloc((size_t)kt_in[layer] * 8 * 512 * 2);
            wrp[layer][d] = (unsigned short*)alloc((size_t)4 * 8 * 512 * 2);
        }
    const int KT0 = (T_ * 256) / 32;     // 2920 K-tiles for dw0
    unsigned short* dw0p = (unsigned short*)alloc((size_t)KT0 * 8 * 512 * 2);
    unsigned short* dw1p = (unsigned short*)alloc((size_t)4 * 8 * 512 * 2);
    unsigned short* dw2p = (unsigned short*)alloc((size_t)4 * 8 * 512 * 2);
    unsigned short* h4 = (unsigned short*)alloc((size_t)B_ * 128 * 2);
    unsigned short* h5 = (unsigned short*)alloc((size_t)B_ * 128 * 2);
    unsigned short* h6 = (unsigned short*)alloc((size_t)B_ * 128 * 2);
    (void)ws_size; (void)in_sizes; (void)n_in; (void)out_size;

    // 1) convert + pad x
    pad_x_kernel<<<(B_ * T_ * 32 + 255) / 256, 256, 0, stream>>>(x, xpad);

    // 2) pack all weights into WMMA B-fragment layout (bf16)
    for (int layer = 0; layer < 3; ++layer)
        for (int d = 0; d < 2; ++d) {
            pack_b_kernel<<<kt_in[layer] * 8, 512, 0, stream>>>(wk[layer][d], in_k[layer],
                                                                wkp[layer][d], 8);
            pack_b_kernel<<<4 * 8, 512, 0, stream>>>(wr[layer][d], 128, wrp[layer][d], 8);
        }
    pack_b_kernel<<<KT0 * 8, 512, 0, stream>>>(dw0, T_ * 256, dw0p, 8);
    pack_b_kernel<<<4 * 8, 512, 0, stream>>>(dw1, 128, dw1p, 8);
    pack_b_kernel<<<4 * 8, 512, 0, stream>>>(dw2, 128, dw2p, 8);

    // 3) three bidirectional RNN layers (fused input-proj + recurrence + relu-concat)
    dim3 sgrid(B_ / 16, 2);
    scan_kernel<1><<<sgrid, 256, 0, stream>>>(xpad,
        wkp[0][0], wrp[0][0], bs[0][0], wkp[0][1], wrp[0][1], bs[0][1], actA);
    scan_kernel<8><<<sgrid, 256, 0, stream>>>(actA,
        wkp[1][0], wrp[1][0], bs[1][0], wkp[1][1], wrp[1][1], bs[1][1], actB);
    scan_kernel<8><<<sgrid, 256, 0, stream>>>(actB,
        wkp[2][0], wrp[2][0], bs[2][0], wkp[2][1], wrp[2][1], bs[2][1], actA);

    // 4) dense head: [1024,93440] @ dw0, then two 128x128 layers
    dense_kernel<<<B_ / 16, 256, 0, stream>>>(actA, T_ * 256, KT0, dw0p, db0, h4);
    dense_kernel<<<B_ / 16, 256, 0, stream>>>(h4, 128, 4, dw1p, db1, h5);
    dense_kernel<<<B_ / 16, 256, 0, stream>>>(h5, 128, 4, dw2p, db2, h6);

    // 5) output GEMV
    head_kernel<<<(B_ + 255) / 256, 256, 0, stream>>>(h6, ow, ob, (float*)d_out);
}
